// ParagraphEncoder_7172595384949
// MI455X (gfx1250) — matrix operand order repaired and checked by reading, not compiled, not run on hardware
//
#include <hip/hip_runtime.h>

// ---------------- problem constants ----------------
constexpr int Bb  = 2;
constexpr int Ss  = 2048;
constexpr int Dd  = 1024;
constexpr int Hh  = 16;
constexpr int HDd = 64;
constexpr int PDd = 2048;
constexpr int NT  = 32;           // B * S / PAR selected rows
constexpr int ROWSTRIDE = 3 * Dd; // qkv row stride = 3072

typedef __bf16 bf16;
typedef __attribute__((ext_vector_type(16))) __bf16 v16bf;
typedef __attribute__((ext_vector_type(8)))  float  v8f;
typedef __attribute__((ext_vector_type(4)))  unsigned u32x4;
typedef __attribute__((ext_vector_type(8)))  int      i32x8;
typedef __attribute__((ext_vector_type(4)))  int      i32x4;

union Frag  { v16bf v; unsigned u[8]; };
union Pack2 { unsigned u; bf16 h[2]; };

// Tensor Data Mover availability (device pass only)
#if defined(__HIP_DEVICE_COMPILE__) && __has_builtin(__builtin_amdgcn_tensor_load_to_lds) && \
    __has_builtin(__builtin_amdgcn_s_wait_tensorcnt)
#define USE_TDM 1
#else
#define USE_TDM 0
#endif

// A-matrix (16x32 bf16) per-lane K index for vgpr r (ISA 7.12.2 table)
__device__ __forceinline__ int kA(int r, int half) {
    return ((r >> 2) << 4) + (half << 3) + ((r & 3) << 1);
}
// B-matrix (32x16 bf16) per-lane K index for vgpr r
__device__ __forceinline__ int kB(int r, int half) {
    return ((r >> 2) << 4) + (half << 3) + ((r & 3) << 1);
}

__device__ __forceinline__ unsigned lds_off(const void* p) {
    // LDS aperture maps the offset into the low 32 bits of the flat address
    return (unsigned)(uintptr_t)p;
}

#if USE_TDM
// Issue a TDM 2-D tile load (bf16 elements): tile_k x tile_rows, row stride = row_stride
// elements, OOB (beyond rem_k/rem_rows) returns zero. D# layout per CDNA5 ISA ch.8.
__device__ __forceinline__ void tdm_load_2d(const bf16* gsrc, unsigned lds_byte_off,
                                            int tile_k, int tile_rows,
                                            long rem_k, long rem_rows,
                                            long row_stride) {
    unsigned long long ga = (unsigned long long)(uintptr_t)gsrc;
    u32x4 g0;
    g0[0] = 1u;                                        // count=1 (valid user descriptor)
    g0[1] = lds_byte_off;                              // lds_addr (bytes)
    g0[2] = (unsigned)(ga & 0xffffffffu);              // global_addr[31:0]
    g0[3] = (unsigned)((ga >> 32) & 0x01ffffffu)       // global_addr[56:32]
          | (2u << 30);                                // type = 2 ("image")
    unsigned long long td0 = (unsigned long long)rem_k;
    unsigned long long td1 = (unsigned long long)rem_rows;
    unsigned long long st0 = (unsigned long long)row_stride;
    i32x8 g1;
    g1[0] = (1 << 16);                                 // data_size=1 (2 bytes); no multicast
    g1[1] = (int)((td0 & 0xffffu) << 16);              // tensor_dim0[15:0]  @ bits 63:48
    g1[2] = (int)(((td0 >> 16) & 0xffffu)              // tensor_dim0[31:16] @ bits 79:64
          | ((td1 & 0xffffu) << 16));                  // tensor_dim1[15:0]  @ bits 95:80
    g1[3] = (int)(((td1 >> 16) & 0xffffu)              // tensor_dim1[31:16]
          | (((unsigned)tile_k & 0xffffu) << 16));     // tile_dim0 @ bits 127:112
    g1[4] = (int)((unsigned)tile_rows & 0xffffu);      // tile_dim1; tile_dim2 = 0
    g1[5] = (int)(st0 & 0xffffffffu);                  // tensor_dim0_stride[31:0]
    g1[6] = (int)((st0 >> 32) & 0xffffu);              // tensor_dim0_stride[47:32]; dim1_stride=0
    g1[7] = 0;
    i32x4 z4 = {0, 0, 0, 0};
#if __clang_major__ >= 23
    i32x8 z8 = {0, 0, 0, 0, 0, 0, 0, 0};
    __builtin_amdgcn_tensor_load_to_lds(g0, g1, z4, z4, z8, 0);
#else
    __builtin_amdgcn_tensor_load_to_lds(g0, g1, z4, z4, 0);
#endif
}
#endif // USE_TDM

// ---------------- f32 -> bf16 convert ----------------
__global__ void cvt_kernel(const float* __restrict__ src, bf16* __restrict__ dst, int n) {
    int i = blockIdx.x * 256 + threadIdx.x;
    int stride = gridDim.x * 256;
    for (; i < n; i += stride) dst[i] = (bf16)src[i];
}

// ---------------- layernorm (rows x D) ----------------
__global__ void ln_kernel(const float* __restrict__ x, const float* __restrict__ g,
                          const float* __restrict__ beta, float* __restrict__ y,
                          bf16* __restrict__ ybf, int D) {
    const int row = blockIdx.x;
    const float* xr = x + (size_t)row * D;
    __shared__ float red[256], red2[256];
    float s = 0.f, s2 = 0.f;
    for (int i = threadIdx.x; i < D; i += 256) { float v = xr[i]; s += v; s2 += v * v; }
    red[threadIdx.x] = s; red2[threadIdx.x] = s2;
    __syncthreads();
    for (int off = 128; off > 0; off >>= 1) {
        if (threadIdx.x < off) { red[threadIdx.x] += red[threadIdx.x + off];
                                 red2[threadIdx.x] += red2[threadIdx.x + off]; }
        __syncthreads();
    }
    float mu  = red[0] / D;
    float var = red2[0] / D - mu * mu;
    float inv = rsqrtf(var + 1e-5f);
    for (int i = threadIdx.x; i < D; i += 256) {
        float v = (xr[i] - mu) * inv * g[i] + beta[i];
        y[(size_t)row * D + i] = v;
        if (ybf) ybf[(size_t)row * D + i] = (bf16)v;
    }
}

// ---------------- boundary prep: seg = cumsum, selidx = nonzero ----------------
__global__ void prep_kernel(const unsigned char* __restrict__ bnd,
                            int* __restrict__ seg, int* __restrict__ selidx) {
    if (blockIdx.x != 0 || threadIdx.x != 0) return;
    int cnt = 0;
    for (int b = 0; b < Bb; ++b) {
        int s = 0;
        for (int t = 0; t < Ss; ++t) {
            int f = bnd[b * Ss + t] ? 1 : 0;
            s += f;
            seg[b * Ss + t] = s;
            if (f && cnt < NT) selidx[cnt++] = b * Ss + t;
        }
    }
}

// ---------------- gather selected rows -> bf16 ----------------
__global__ void gather_kernel(const float* __restrict__ x2, const int* __restrict__ selidx,
                              bf16* __restrict__ sel) {
    const int i = blockIdx.x;
    const int row = selidx[i];
    for (int d = threadIdx.x; d < Dd; d += 256)
        sel[(size_t)i * Dd + d] = (bf16)x2[(size_t)row * Dd + d];
}

// ---------------- WMMA GEMM, TDM/LDS staged:  C[M,N] = A[M,K] @ W[N,K]^T + bias ----------------
// 256 threads (8 waves); block tile 64(M) x 128(N); K-slab 32, double buffered via TDM.
// EPI: 0 = store bf16, 1 = +resid f32, 2 = gelu bf16, 3 = f32
template <int EPI>
__global__ void __launch_bounds__(256)
gemm_wmma(const bf16* __restrict__ A, const bf16* __restrict__ W,
          const float* __restrict__ bias, float* __restrict__ Cf,
          bf16* __restrict__ Cb, const float* __restrict__ resid,
          int M, int N, int K) {
    const int tid  = threadIdx.x;
    const int wave = tid >> 5, lane = tid & 31;
    const int half = lane >> 4, lr = lane & 15;
    const int wm = wave >> 2;          // 0..1 : M sub-block
    const int wn = wave & 3;           // 0..3 : N sub-block
    const int n0 = blockIdx.x * 128;
    const int m0 = blockIdx.y * 64;

    __shared__ bf16 sA[2][64 * 32];    //  8 KB
    __shared__ bf16 sW[2][128 * 32];   // 16 KB

    auto load_slab = [&](int buf, int k0) {
#if USE_TDM
        if (wave == 0) {
            tdm_load_2d(A + (size_t)m0 * K + k0, lds_off(&sA[buf][0]),
                        32, 64, (long)(K - k0), (long)M - m0, (long)K);
            tdm_load_2d(W + (size_t)n0 * K + k0, lds_off(&sW[buf][0]),
                        32, 128, (long)(K - k0), (long)N - n0, (long)K);
        }
#else
        for (int i = tid; i < 64 * 16; i += 256) {
            int row = i >> 4, c = (i & 15) << 1;
            unsigned v = (m0 + row < M)
                ? *(const unsigned*)(A + (size_t)(m0 + row) * K + k0 + c) : 0u;
            *(unsigned*)&sA[buf][row * 32 + c] = v;
        }
        for (int i = tid; i < 128 * 16; i += 256) {
            int row = i >> 4, c = (i & 15) << 1;
            unsigned v = *(const unsigned*)(W + (size_t)(n0 + row) * K + k0 + c);
            *(unsigned*)&sW[buf][row * 32 + c] = v;
        }
#endif
    };

    v8f acc[2][2] = {};
    load_slab(0, 0);
#if USE_TDM
    if (wave == 0) __builtin_amdgcn_s_wait_tensorcnt(0);
#endif
    __syncthreads();

    int cur = 0;
    for (int k0 = 0; k0 < K; k0 += 32, cur ^= 1) {
        if (k0 + 32 < K) load_slab(cur ^ 1, k0 + 32);       // async prefetch of next slab
        Frag a[2], b[2];
#pragma unroll
        for (int r = 0; r < 8; ++r) {
            const int ka = kA(r, half), kb = kB(r, half);
            a[0].u[r] = *(const unsigned*)&sA[cur][(wm * 32 + lr)      * 32 + ka];
            a[1].u[r] = *(const unsigned*)&sA[cur][(wm * 32 + 16 + lr) * 32 + ka];
            b[0].u[r] = *(const unsigned*)&sW[cur][(wn * 32 + lr)      * 32 + kb];
            b[1].u[r] = *(const unsigned*)&sW[cur][(wn * 32 + 16 + lr) * 32 + kb];
        }
#pragma unroll
        for (int mi = 0; mi < 2; ++mi)
#pragma unroll
            for (int ni = 0; ni < 2; ++ni)
                acc[mi][ni] = __builtin_amdgcn_wmma_f32_16x16x32_bf16(
                    false, a[mi].v, false, b[ni].v, (short)0, acc[mi][ni], false, false);
#if USE_TDM
        if (wave == 0) __builtin_amdgcn_s_wait_tensorcnt(0);
#endif
        __syncthreads();
    }

#pragma unroll
    for (int mi = 0; mi < 2; ++mi)
#pragma unroll
        for (int ni = 0; ni < 2; ++ni)
#pragma unroll
            for (int r = 0; r < 8; ++r) {
                const int row = m0 + wm * 32 + mi * 16 + r + (half << 3);
                const int col = n0 + wn * 32 + ni * 16 + lr;
                if (row >= M) continue;
                float v = acc[mi][ni][r] + bias[col];
                const size_t idx = (size_t)row * N + col;
                if constexpr (EPI == 0) {
                    Cb[idx] = (bf16)v;
                } else if constexpr (EPI == 1) {
                    Cf[idx] = v + resid[idx];
                } else if constexpr (EPI == 2) {
                    float gl = 0.5f * v * (1.0f + erff(v * 0.70710678118654752f));
                    Cb[idx] = (bf16)gl;
                } else {
                    Cf[idx] = v;
                }
            }
}

// ---------------- block-bias attention (flash-style, chunk = 1024 keys) ----------------
__global__ void __launch_bounds__(128)
attn_kernel(const bf16* __restrict__ qkv, const int* __restrict__ seg,
            bf16* __restrict__ ctx) {
    const int qt = blockIdx.x, h = blockIdx.y, b = blockIdx.z;
    const int tid = threadIdx.x, wave = tid >> 5, lane = tid & 31;
    const int half = lane >> 4, lr = lane & 15;

    __shared__ bf16  sc[16][1024 + 8];         // raw scores, then P=exp(s-m) in place
    __shared__ bf16  sV[4][2][32 * 16];        // per-wave double-buffered V sub-slab (8 KB)
    __shared__ float s_m[16], s_l[16], s_scale[16];

    const bf16* Q  = qkv + ((size_t)(b * Ss + qt * 16)) * ROWSTRIDE + h * HDd;
    const bf16* Kp = qkv + (size_t)(b * Ss) * ROWSTRIDE + Dd + h * HDd;
    const bf16* Vp = qkv + (size_t)(b * Ss) * ROWSTRIDE + 2 * Dd + h * HDd;
    const int* segb = seg + b * Ss;
    const int segLast = segb[Ss - 1];

    Frag qa[2];
#pragma unroll
    for (int r = 0; r < 8; ++r) {
        const int ka = kA(r, half);
        qa[0].u[r] = *(const unsigned*)(Q + (size_t)lr * ROWSTRIDE + ka);
        qa[1].u[r] = *(const unsigned*)(Q + (size_t)lr * ROWSTRIDE + 32 + ka);
    }
    int segq[8];
#pragma unroll
    for (int r = 0; r < 8; ++r) segq[r] = segb[qt * 16 + r + (half << 3)];

    if (tid < 16) { s_m[tid] = -1e30f; s_l[tid] = 0.f; }
    v8f acc = {};
    __syncthreads();

    for (int c = 0; c < 2; ++c) {
        // ---- pass 1: score tiles for keys [c*1024, c*1024+1024) ----
        for (int kt = wave; kt < 64; kt += 4) {
            const int k0tok = c * 1024 + kt * 16;
            Frag kb[2];
#pragma unroll
            for (int r = 0; r < 8; ++r) {
                const int kbk = kB(r, half);
                const bf16* kp = Kp + (size_t)(k0tok + lr) * ROWSTRIDE + kbk;
                kb[0].u[r] = *(const unsigned*)(kp);
                kb[1].u[r] = *(const unsigned*)(kp + 32);
            }
            v8f s = {};
            s = __builtin_amdgcn_wmma_f32_16x16x32_bf16(false, qa[0].v, false, kb[0].v, (short)0, s, false, false);
            s = __builtin_amdgcn_wmma_f32_16x16x32_bf16(false, qa[1].v, false, kb[1].v, (short)0, s, false, false);
            const int sk = segb[k0tok + lr];
#pragma unroll
            for (int r = 0; r < 8; ++r) {
                const float biasv = (segq[r] == sk && segq[r] != segLast) ? 1.0f : 0.0f;
                sc[r + (half << 3)][kt * 16 + lr] = (bf16)(s[r] * 0.125f + biasv);
            }
        }
        __syncthreads();

        // ---- pass 2: row stats + in-place P = exp(s - m_new) ----
        {
            const int row = tid >> 3, sub = tid & 7;
            float mx = -1e30f;
            for (int j = sub; j < 1024; j += 8) mx = fmaxf(mx, (float)sc[row][j]);
            mx = fmaxf(mx, __shfl_xor(mx, 1));
            mx = fmaxf(mx, __shfl_xor(mx, 2));
            mx = fmaxf(mx, __shfl_xor(mx, 4));
            const float mold = s_m[row];              // read BEFORE lane-0 updates it
            const float mnew = fmaxf(mold, mx);
            float sum = 0.f;
            for (int j = sub; j < 1024; j += 8) {
                float e = expf((float)sc[row][j] - mnew);
                sum += e;
                sc[row][j] = (bf16)e;                 // exponentiate once, in place
            }
            sum += __shfl_xor(sum, 1);
            sum += __shfl_xor(sum, 2);
            sum += __shfl_xor(sum, 4);
            if (sub == 0) {
                const float sca = expf(mold - mnew);
                s_scale[row] = sca;
                s_l[row] = s_l[row] * sca + sum;
                s_m[row] = mnew;
            }
        }
        __syncthreads();

        // ---- pass 3: O = O*scale + P @ V  (each wave owns 16 hd cols) ----
#pragma unroll
        for (int r = 0; r < 8; ++r) acc[r] *= s_scale[r + (half << 3)];
#if USE_TDM
        tdm_load_2d(Vp + (size_t)(c * 1024) * ROWSTRIDE + wave * 16,
                    lds_off(&sV[wave][0][0]), 16, 32, 16, 32, (long)ROWSTRIDE);
        __builtin_amdgcn_s_wait_tensorcnt(0);
#endif
        int vcur = 0;
        for (int kk = 0; kk < 1024; kk += 32) {
#if USE_TDM
            if (kk + 32 < 1024)
                tdm_load_2d(Vp + (size_t)(c * 1024 + kk + 32) * ROWSTRIDE + wave * 16,
                            lds_off(&sV[wave][vcur ^ 1][0]), 16, 32, 16, 32, (long)ROWSTRIDE);
#endif
            Frag pa, vb;
#pragma unroll
            for (int r = 0; r < 8; ++r) {
                pa.u[r] = *(const unsigned*)&sc[lr][kk + kA(r, half)];
                const int kbk = kB(r, half);
                Pack2 vk;
#if USE_TDM
                vk.h[0] = sV[wave][vcur][kbk * 16 + lr];
                vk.h[1] = sV[wave][vcur][(kbk + 1) * 16 + lr];
#else
                const bf16* vp = Vp + (size_t)(c * 1024 + kk + kbk) * ROWSTRIDE + wave * 16 + lr;
                vk.h[0] = vp[0]; vk.h[1] = vp[ROWSTRIDE];
#endif
                vb.u[r] = vk.u;
            }
            acc = __builtin_amdgcn_wmma_f32_16x16x32_bf16(false, pa.v, false, vb.v, (short)0, acc, false, false);
#if USE_TDM
            __builtin_amdgcn_s_wait_tensorcnt(0);
            vcur ^= 1;
#endif
        }
        __syncthreads();
    }

#pragma unroll
    for (int r = 0; r < 8; ++r) {
        const int m = r + (half << 3);
        const float o = acc[r] / s_l[m];
        ctx[((size_t)(b * Ss + qt * 16 + m)) * Dd + h * HDd + wave * 16 + lr] = (bf16)o;
    }
}

// ---------------- host-side orchestration ----------------
extern "C" void kernel_launch(void* const* d_in, const int* in_sizes, int n_in,
                              void* d_out, int out_size, void* d_ws, size_t ws_size,
                              hipStream_t stream) {
    (void)in_sizes; (void)n_in; (void)out_size; (void)ws_size;

    const float*         x_in   = (const float*)d_in[0];
    const unsigned char* bnd    = (const unsigned char*)d_in[1];
    const float*         w_in   = (const float*)d_in[2];   // (3D, D)
    const float*         b_in   = (const float*)d_in[3];
    const float*         w_out  = (const float*)d_in[4];   // (D, D)
    const float*         b_out  = (const float*)d_in[5];
    const float*         ln1_g  = (const float*)d_in[6];
    const float*         ln1_b  = (const float*)d_in[7];
    const float*         w_ff1  = (const float*)d_in[8];   // (PD, D)
    const float*         b_ff1  = (const float*)d_in[9];
    const float*         w_ff2  = (const float*)d_in[10];  // (PD, PD)
    const float*         b_ff2  = (const float*)d_in[11];
    const float*         ln2_g  = (const float*)d_in[12];
    const float*         ln2_b  = (const float*)d_in[13];
    float* out = (float*)d_out;

    const int M = Bb * Ss;  // 4096

    char* p = (char*)d_ws;
    auto alloc = [&](size_t bytes) { void* r = (void*)p; p += (bytes + 255) & ~(size_t)255; return r; };

    float* x_ln    = (float*)alloc((size_t)M * Dd * 4);
    bf16*  x_bf    = (bf16*) alloc((size_t)M * Dd * 2);
    bf16*  win_bf  = (bf16*) alloc((size_t)3 * Dd * Dd * 2);
    bf16*  wout_bf = (bf16*) alloc((size_t)Dd * Dd * 2);
    bf16*  wff1_bf = (bf16*) alloc((size_t)PDd * Dd * 2);
    bf16*  wff2_bf = (bf16*) alloc((size_t)PDd * PDd * 2);
    bf16*  qkv_bf  = (bf16*) alloc((size_t)M * 3 * Dd * 2);
    bf16*  ctx_bf  = (bf16*) alloc((size_t)M * Dd * 2);
    float* x2      = (float*)alloc((size_t)M * Dd * 4);
    bf16*  sel_bf  = (bf16*) alloc((size_t)NT * Dd * 2);
    bf16*  h1_bf   = (bf16*) alloc((size_t)NT * PDd * 2);
    float* h2      = (float*)alloc((size_t)NT * PDd * 4);
    int*   seg     = (int*)  alloc((size_t)Bb * Ss * 4);
    int*   selidx  = (int*)  alloc((size_t)NT * 4);

    prep_kernel<<<1, 32, 0, stream>>>(bnd, seg, selidx);

    auto cvt = [&](const float* s, bf16* d, int n) {
        int g = (n + 255) / 256; if (g > 4096) g = 4096;
        cvt_kernel<<<g, 256, 0, stream>>>(s, d, n);
    };
    cvt(w_in,  win_bf,  3 * Dd * Dd);
    cvt(w_out, wout_bf, Dd * Dd);
    cvt(w_ff1, wff1_bf, PDd * Dd);
    cvt(w_ff2, wff2_bf, PDd * PDd);

    ln_kernel<<<M, 256, 0, stream>>>(x_in, ln1_g, ln1_b, x_ln, x_bf, Dd);

    // QKV projection
    gemm_wmma<0><<<dim3(3 * Dd / 128, M / 64), 256, 0, stream>>>(
        x_bf, win_bf, b_in, nullptr, qkv_bf, nullptr, M, 3 * Dd, Dd);

    // block-bias attention
    attn_kernel<<<dim3(Ss / 16, Hh, Bb), 128, 0, stream>>>(qkv_bf, seg, ctx_bf);

    // out projection + residual
    gemm_wmma<1><<<dim3(Dd / 128, M / 64), 256, 0, stream>>>(
        ctx_bf, wout_bf, b_out, x2, nullptr, x_ln, M, Dd, Dd);

    gather_kernel<<<NT, 256, 0, stream>>>(x2, selidx, sel_bf);

    // FF1 + exact gelu
    gemm_wmma<2><<<dim3(PDd / 128, 1), 256, 0, stream>>>(
        sel_bf, wff1_bf, b_ff1, nullptr, h1_bf, nullptr, NT, PDd, Dd);

    // FF2
    gemm_wmma<3><<<dim3(PDd / 128, 1), 256, 0, stream>>>(
        h1_bf, wff2_bf, b_ff2, h2, nullptr, nullptr, NT, PDd, PDd);

    ln_kernel<<<NT, 256, 0, stream>>>(h2, ln2_g, ln2_b, out, nullptr, PDd);
}
